// NonIsotropicNCC_29274497089744
// MI455X (gfx1250) — matrix operand order repaired and checked by reading, not compiled, not run on hardware
//
#include <hip/hip_runtime.h>

// ---------------------------------------------------------------------------
// NonIsotropicNCC on MI455X (gfx1250, wave32).
//
// sym_block(x,w,b) = relu(conv4d(x,w)+b) + relu(conv4d(x, swap(w))+b),
// where swap(w) exchanges the (k1a,k1b) and (k2a,k2b) kernel axes
// (== perm(f(perm(x))): no tensor transposes needed).
//
// Ci=16 layers (88% of FLOPs): implicit GEMM on V_WMMA_F32_16X16X32_BF16
// with a 3-term bf16 split (hi*hi + hi*lo + lo*hi ~= fp32 product).
//   Per wave: 16(d4) x 16(N) x 5(d3-tiles) outputs, 5 accumulators, so each
//   B-operand register load is reused 5x (cuts L2 weight traffic 5x).
//   K-chunk = 32 = 2 l-taps x 16 ci (K = lsub*16 + ci -> per-VGPR bf16 pairs
//   are adjacent ci -> aligned ds_load_b128 operand fetch from an LDS slab
//   of 9 e3-rows staged once per (i,j) tap pair).
// Ci=1 layer: V_WMMA_F32_16X16X4_F32 implicit GEMM (small, keep fp32).
// ---------------------------------------------------------------------------

#define SS  25
#define SS2 625
#define SS3 15625
#define SS4 390625

typedef __attribute__((ext_vector_type(2)))  float  v2f;
typedef __attribute__((ext_vector_type(8)))  float  v8f;
typedef __attribute__((ext_vector_type(16))) __bf16 v16bf;

union ABu { uint4 u[2]; v16bf v; };

__device__ __forceinline__ float hi_part(float a) {
    return __builtin_bit_cast(float, __builtin_bit_cast(unsigned, a) & 0xffff0000u);
}
// pack bf16(a) into [15:0], bf16(b) into [31:16]  (mantissa truncation)
__device__ __forceinline__ unsigned trunc_pack(float a, float b) {
    return (__builtin_bit_cast(unsigned, a) >> 16) |
           (__builtin_bit_cast(unsigned, b) & 0xffff0000u);
}

// ---------------------------------------------------------------------------
// f32 weight prep for the Ci=1 layer (WMMA-B layout for 16x16x4 f32):
//   flat = ((((ijk*2+ch)*2+kp)*2+u)*16+n)*2+q (u = f bit 5, ignored), padded l tap ll = ch*4+kp*2+q
// ---------------------------------------------------------------------------
__global__ void prep_bmat_f32(const float* __restrict__ w, float* __restrict__ bmat,
                              int co, int ka, int kb, int kc, int kd, int total)
{
    int f = blockIdx.x * blockDim.x + threadIdx.x;
    if (f >= total) return;

    int q  = f & 1;
    int n  = (f >> 1) & 15;
    int kp = (f >> 6) & 1;
    int r  = f >> 7;
    int ch  = r & 1;
    int ijk = r >> 1;
    int A = ijk / 25, Bc = (ijk / 5) % 5, C = ijk % 5;
    int D = ch * 4 + kp * 2 + q;

    int oa = 2 - ka / 2, ob = 2 - kb / 2, oc = 2 - kc / 2, od = 2 - kd / 2;
    float val = 0.f;
    if (n < co) {
        int i = A - oa, j = Bc - ob, k = C - oc, l = D - od;
        if (i >= 0 && i < ka && j >= 0 && j < kb && k >= 0 && k < kc && l >= 0 && l < kd)
            val = w[(((n * ka + i) * kb + j) * kc + k) * kd + l];           // Ci==1
    } else if (n < 2 * co) {
        int mm = n - co;
        int k = A - oc, l = Bc - od, i = C - oa, j = D - ob;
        if (i >= 0 && i < ka && j >= 0 && j < kb && k >= 0 && k < kc && l >= 0 && l < kd)
            val = w[(((mm * ka + i) * kb + j) * kc + k) * kd + l];
    }
    bmat[f] = val;
}

// ---------------------------------------------------------------------------
// bf16 hi/lo weight prep for Ci=16 layers, pre-swizzled to the WMMA-B lane
// layout: bf16 element flat = (((ijk*3+cidx)*2+lsub)*16 + n)*16 + ci,
// where the conv tap is l = cidx*2 + lsub (l==5 -> zero pad).
// ---------------------------------------------------------------------------
__global__ void prep_bmat_bf16(const float* __restrict__ w,
                               unsigned short* __restrict__ bmh,
                               unsigned short* __restrict__ bml,
                               int co, int ka, int kb, int kc, int kd, int total)
{
    int f = blockIdx.x * blockDim.x + threadIdx.x;
    if (f >= total) return;

    const int Ci = 16;
    int ci   = f & 15;
    int n    = (f >> 4) & 15;
    int lsub = (f >> 8) & 1;
    int rest = f >> 9;
    int cidx = rest % 3;
    int ijk  = rest / 3;
    int A = ijk / 25, Bc = (ijk / 5) % 5, C = ijk % 5;
    int D = cidx * 2 + lsub;

    int oa = 2 - ka / 2, ob = 2 - kb / 2, oc = 2 - kc / 2, od = 2 - kd / 2;
    float val = 0.f;
    if (D < 5) {
        if (n < co) {
            int i = A - oa, j = Bc - ob, k = C - oc, l = D - od;
            if (i >= 0 && i < ka && j >= 0 && j < kb && k >= 0 && k < kc && l >= 0 && l < kd)
                val = w[((((n * Ci + ci) * ka + i) * kb + j) * kc + k) * kd + l];
        } else if (n < 2 * co) {
            int mm = n - co;
            int k = A - oc, l = Bc - od, i = C - oa, j = D - ob;
            if (i >= 0 && i < ka && j >= 0 && j < kb && k >= 0 && k < kc && l >= 0 && l < kd)
                val = w[((((mm * Ci + ci) * ka + i) * kb + j) * kc + k) * kd + l];
        }
    }
    float lo = val - hi_part(val);
    bmh[f] = (unsigned short)(__builtin_bit_cast(unsigned, val) >> 16);
    bml[f] = (unsigned short)(__builtin_bit_cast(unsigned, lo)  >> 16);
}

// ---------------------------------------------------------------------------
// Ci=16 fused symmetric-block conv: bf16 split-precision WMMA 16x16x32.
// One wave computes a 16(d4) x 16(N) x 5(d3) output block.
// ---------------------------------------------------------------------------
__global__ __launch_bounds__(32)
void conv_sym_bf16(const float* __restrict__ x,
                   const unsigned short* __restrict__ bmh,
                   const unsigned short* __restrict__ bml,
                   const float* __restrict__ bias, float* __restrict__ out,
                   int Cout, int co_base, int co)
{
    // packed bf16 pairs (ci even in [15:0]), 9 e3-rows x 21 e4 x 8 ci-words
    __shared__ unsigned ldsH[9][21][8];
    __shared__ unsigned ldsL[9][21][8];
    const int Ci = 16;

    int lane = threadIdx.x;
    int m    = lane & 15;
    int hi   = lane >> 4;

    int wg = blockIdx.x;
    int t4 = wg & 1; wg >>= 1;
    int dt = wg % 5; wg /= 5;
    int d2 = wg % SS; wg /= SS;
    int d1 = wg % SS; wg /= SS;
    int b  = wg;
    int d4base = t4 * 16;
    int d3base = dt * 5;

    v8f acc[5] = {};

    for (int ab = 0; ab < 25; ++ab) {               // (i,j) tap pair
        int A = ab / 5, Bc = ab % 5;
        int e1 = d1 + A - 2, e2 = d2 + Bc - 2;
        bool sv = (unsigned)e1 < SS && (unsigned)e2 < SS;
        int base = (b * Ci) * SS4 + e1 * SS3 + e2 * SS2;

        __syncthreads();
        // stage hi/lo slab: 9 x 21 x 8 packed words each (1512 words/array)
        #pragma unroll 4
        for (int it = 0; it < 48; ++it) {
            int s = lane + it * 32;
            if (s < 1512) {
                int r = s / 168, rem = s - r * 168;
                int p = rem >> 3, c = rem & 7;
                int e3 = d3base - 2 + r;
                int e4 = d4base - 2 + p;
                float v0 = 0.f, v1 = 0.f;
                if (sv && (unsigned)e3 < SS && (unsigned)e4 < SS) {
                    int idx = base + e3 * SS + e4;
                    v0 = x[idx + (2 * c)     * SS4];
                    v1 = x[idx + (2 * c + 1) * SS4];
                }
                ldsH[r][p][c] = trunc_pack(v0, v1);
                ldsL[r][p][c] = trunc_pack(v0 - hi_part(v0), v1 - hi_part(v1));
            }
        }
        __syncthreads();

        // prefetch next slab's rows (gfx1250 global_prefetch_b8)
        if (ab + 1 < 25) {
            int A2 = (ab + 1) / 5, B2 = (ab + 1) % 5;
            int f1 = d1 + A2 - 2, f2 = d2 + B2 - 2;
            if ((unsigned)f1 < SS && (unsigned)f2 < SS)
                __builtin_prefetch(x + (b * Ci) * SS4 + f1 * SS3 + f2 * SS2
                                     + (d3base > 2 ? (d3base - 2) * SS : 0), 0, 1);
        }

        for (int C = 0; C < 5; ++C) {               // third kernel axis tap
            int ijk = ab * 5 + C;
            #pragma unroll
            for (int cidx = 0; cidx < 3; ++cidx) {  // K chunks (l pairs)
                // B operand: pre-swizzled weights, held in regs across 5 tiles
                ABu Bh, Bl;
                size_t off = ((size_t)((ijk * 6 + cidx * 2 + hi) * 16 + m)) * 16;
                Bh.u[0] = *(const uint4*)(bmh + off);
                Bh.u[1] = *(const uint4*)(bmh + off + 8);
                Bl.u[0] = *(const uint4*)(bml + off);
                Bl.u[1] = *(const uint4*)(bml + off + 8);

                int p0 = m + cidx * 2, p1 = p0 + 1; // lsub = 0 / 1 positions
                #pragma unroll
                for (int t = 0; t < 5; ++t) {       // d3 tiles
                    int r = t + C;                  // slab row
                    ABu Ah, Al;
                    Ah.u[0] = *(const uint4*)&ldsH[r][p0 + hi][0];
                    Ah.u[1] = *(const uint4*)&ldsH[r][p0 + hi][4];
                    Al.u[0] = *(const uint4*)&ldsL[r][p0 + hi][0];
                    Al.u[1] = *(const uint4*)&ldsL[r][p0 + hi][4];

                    // 3-term split: hi*hi + hi*lo + lo*hi
                    acc[t] = __builtin_amdgcn_wmma_f32_16x16x32_bf16(
                        false, Ah.v, false, Bh.v, (short)0, acc[t], false, false);
                    acc[t] = __builtin_amdgcn_wmma_f32_16x16x32_bf16(
                        false, Ah.v, false, Bl.v, (short)0, acc[t], false, false);
                    acc[t] = __builtin_amdgcn_wmma_f32_16x16x32_bf16(
                        false, Al.v, false, Bh.v, (short)0, acc[t], false, false);
                }
            }
        }
    }

    // Epilogue: +bias, relu both branches, pair columns n and n+co, store.
    int n = m;
    int bidx = (n < co) ? n : ((n < 2 * co) ? n - co : 0);
    float bval = (n < 2 * co) ? bias[bidx] : 0.f;
    #pragma unroll
    for (int t = 0; t < 5; ++t) {
        int d3 = d3base + t;
        #pragma unroll
        for (int v = 0; v < 8; ++v) {
            float r = fmaxf(acc[t][v] + bval, 0.f);
            float p = __shfl_xor(r, co, 32);
            if (n < co) {
                int d4 = d4base + v + 8 * hi;
                if (d4 < SS)
                    out[(b * Cout + co_base + n) * SS4 + d1 * SS3 + d2 * SS2 + d3 * SS + d4]
                        = r + p;
            }
        }
    }
}

// ---------------------------------------------------------------------------
// Ci=1 fused symmetric-block conv: f32 WMMA 16x16x4 (layer 0 only, ~12% of
// the FLOPs -- fp32 matrix path is plenty here).
// ---------------------------------------------------------------------------
__global__ __launch_bounds__(32)
void conv_sym_f32(const float* __restrict__ x, const float* __restrict__ bmat,
                  const float* __restrict__ bias, float* __restrict__ out,
                  int Cout, int co_base, int co)
{
    __shared__ float lds[24];

    int lane = threadIdx.x;
    int m    = lane & 15;
    int hi   = lane >> 4;

    int wg = blockIdx.x;
    int t4 = wg & 1;  wg >>= 1;
    int d3 = wg % SS; wg /= SS;
    int d2 = wg % SS; wg /= SS;
    int d1 = wg % SS; wg /= SS;
    int b  = wg;
    int d4base = t4 * 16;

    v8f acc = {};

    for (int ijk = 0; ijk < 125; ++ijk) {
        int A  = ijk / 25, Bc = (ijk / 5) % 5, C = ijk % 5;
        int e1 = d1 + A - 2, e2 = d2 + Bc - 2, e3 = d3 + C - 2;
        bool sval = (unsigned)e1 < SS && (unsigned)e2 < SS && (unsigned)e3 < SS;
        int base = b * SS4 + e1 * SS3 + e2 * SS2 + e3 * SS;

        __syncthreads();
        if (lane < 23) {
            int e4 = d4base - 2 + lane;
            float v = 0.f;
            if (sval && (unsigned)e4 < SS) v = x[base + e4];
            lds[lane] = v;
        }
        __syncthreads();

        const float* bp = bmat + (size_t)ijk * 128;        // 2*2*16*2
        #pragma unroll
        for (int ch = 0; ch < 2; ++ch) {
            int ll = ch * 4 + hi * 2;                      // padded l tap
            v2f a;
            a.x = lds[m + ll];
            a.y = lds[m + ll + 1];
            v2f bb = *(const v2f*)(bp + ((ch * 2 + hi) * 16 + m) * 2);
            acc = __builtin_amdgcn_wmma_f32_16x16x4_f32(
                false, a, false, bb, (short)0, acc, false, false);
        }
    }

    int n = m;
    int bidx = (n < co) ? n : ((n < 2 * co) ? n - co : 0);
    float bval = (n < 2 * co) ? bias[bidx] : 0.f;
    #pragma unroll
    for (int v = 0; v < 8; ++v) {
        float r = fmaxf(acc[v] + bval, 0.f);
        float p = __shfl_xor(r, co, 32);
        if (n < co) {
            int d4 = d4base + v + 8 * hi;
            if (d4 < SS)
                out[(b * Cout + co_base + n) * SS4 + d1 * SS3 + d2 * SS2 + d3 * SS + d4]
                    = r + p;
        }
    }
}

// ---------------------------------------------------------------------------
extern "C" void kernel_launch(void* const* d_in, const int* in_sizes, int n_in,
                              void* d_out, int out_size, void* d_ws, size_t ws_size,
                              hipStream_t stream)
{
    const float* x   = (const float*)d_in[0];
    const float* w00 = (const float*)d_in[1];
    const float* b00 = (const float*)d_in[2];
    const float* w01 = (const float*)d_in[3];
    const float* b01 = (const float*)d_in[4];
    const float* w10 = (const float*)d_in[5];
    const float* b10 = (const float*)d_in[6];
    const float* w11 = (const float*)d_in[7];
    const float* b11 = (const float*)d_in[8];
    const float* w2  = (const float*)d_in[9];
    const float* b2  = (const float*)d_in[10];
    float* out = (float*)d_out;

    float* ws = (float*)d_ws;
    const size_t NY = (size_t)2 * 16 * SS4;       // 12.5M floats
    float* y1   = ws; ws += NY;
    float* y2   = ws; ws += NY;
    float* bm0a = ws; ws += 16000;                // f32 layout: 125*2*2*16*2
    float* bm0b = ws; ws += 16000;

    const int NB16 = 125 * 3 * 2 * 16 * 16;       // 192000 bf16 elements
    unsigned short* us = (unsigned short*)ws;
    unsigned short* bm1ah = us; us += NB16;
    unsigned short* bm1al = us; us += NB16;
    unsigned short* bm1bh = us; us += NB16;
    unsigned short* bm1bl = us; us += NB16;
    unsigned short* bm2h  = us; us += NB16;
    unsigned short* bm2l  = us; us += NB16;

    // weight prep (tiny)
    prep_bmat_f32 <<<(16000 + 255) / 256, 256, 0, stream>>>(w00, bm0a, 8, 3, 3, 5, 5, 16000);
    prep_bmat_f32 <<<(16000 + 255) / 256, 256, 0, stream>>>(w01, bm0b, 8, 5, 5, 5, 5, 16000);
    prep_bmat_bf16<<<(NB16 + 255) / 256, 256, 0, stream>>>(w10, bm1ah, bm1al, 8, 3, 3, 5, 5, NB16);
    prep_bmat_bf16<<<(NB16 + 255) / 256, 256, 0, stream>>>(w11, bm1bh, bm1bl, 8, 5, 5, 5, 5, NB16);
    prep_bmat_bf16<<<(NB16 + 255) / 256, 256, 0, stream>>>(w2,  bm2h,  bm2l,  1, 5, 5, 5, 5, NB16);

    const int NWG_F32  = 2 * SS * SS * SS * 2;    // 62500 waves (layer 0)
    const int NWG_BF16 = 2 * SS * SS * 5 * 2;     // 12500 waves (5 d3-tiles each)

    // layer 0 (Ci=1, f32 WMMA): two sym blocks -> y1 channels [0,8) / [8,16)
    conv_sym_f32 <<<NWG_F32, 32, 0, stream>>>(x,  bm0a, b00, y1, 16, 0, 8);
    conv_sym_f32 <<<NWG_F32, 32, 0, stream>>>(x,  bm0b, b01, y1, 16, 8, 8);
    // layer 1 (Ci=16, bf16-split WMMA): two sym blocks -> y2
    conv_sym_bf16<<<NWG_BF16, 32, 0, stream>>>(y1, bm1ah, bm1al, b10, y2, 16, 0, 8);
    conv_sym_bf16<<<NWG_BF16, 32, 0, stream>>>(y1, bm1bh, bm1bl, b11, y2, 16, 8, 8);
    // layer 2 (Ci=16, Co=1) -> out
    conv_sym_bf16<<<NWG_BF16, 32, 0, stream>>>(y2, bm2h,  bm2l,  b2, out, 1, 0, 1);
}
